// cnnToLSTMCustom_10307921511186
// MI455X (gfx1250) — matrix operand order repaired and checked by reading, not compiled, not run on hardware
//
#include <hip/hip_runtime.h>
#include <hip/hip_bf16.h>
#include <math.h>

// ---------------- types ----------------
typedef __attribute__((ext_vector_type(16))) __bf16 v16bf;
typedef __attribute__((ext_vector_type(8)))  float  v8f;

#define B_   16
#define L_   256
#define D_   300
#define KP_  320      // K padded to 10*32 for WMMA
#define KT_IH 10      // K tiles (32-wide) for input projection
#define KT_HH 16      // K tiles for recurrence (512/32)
#define T_   512      // OUT_LEN
#define H_   512
#define G4_  2048     // 4*H
#define NT_  128      // N tiles (2048/16)

__device__ __forceinline__ float sigmoidf_(float x){ return 1.f/(1.f+expf(-x)); }

// A-fragment sub-pattern within a 32-wide K chunk: k -> (hi, e)
__device__ __forceinline__ void a_pat(int km, int& hi, int& e){
  hi = (km >> 3) & 1;
  e  = (km & 7) + ((km >= 16) ? 8 : 0);
}

// ---------------- scatter inverse maps (pos -> conv src index, or -1) ------
__device__ __forceinline__ int inv_k2(int p){
  if(p==0) return -1;
  int i = (p&1) ? (p-1)>>1 : (p-2)>>1;
  return (i>=0 && i<255) ? i : -1;
}
__device__ __forceinline__ int inv_k4(int p){
  int r=p&3, i;
  if(r==1) i=(p-1)>>2; else if(r==3) i=(p-3)>>2;
  else if(r==0) i=(p-4)>>2; else i=(p-6)>>2;
  return (i>=0 && i<=126) ? i : -1;
}
__device__ __forceinline__ int inv_k3(int p){
  if(!(p&1)) return -1;
  int r=p%6, i;
  if(r==5) i=(p+1)/6; else if(r==1) i=(p-1)/6; else i=(p-3)/6;
  return (i>=0 && i<=85) ? i : -1;
}
__device__ __forceinline__ int inv_k6(int p){
  int r=p%6, i;
  switch(r){
    case 5: i=(p+1)/6; break; case 1: i=(p-1)/6; break; case 3: i=(p-3)/6; break;
    case 4: i=(p-4)/6; break; case 0: i=(p-6)/6; break; default: i=(p-8)/6; break;
  }
  return (i>=0 && i<=84) ? i : -1;
}
__device__ __forceinline__ int inv_k5(int p){
  int r=p%6, i;
  switch(r){
    case 2: i=(p+4)/6; break; case 4: i=(p+2)/6; break; case 1: i=(p-1)/6; break;
    case 3: i=(p-3)/6; break; case 5: i=(p-5)/6; break; default: return -1;
  }
  return (i>=0 && i<=85) ? i : -1;
}

// ---------------- kernel 1: embedding gather -> emb[b][t][c] f32 ----------
__global__ void k_embed(const int* __restrict__ tok, const float* __restrict__ tbl,
                        float* __restrict__ emb){
  int id = blockIdx.x*blockDim.x + threadIdx.x;
  if(id >= B_*L_*D_) return;
  int c  = id % D_;
  int bt = id / D_;                 // b*256+t
  emb[id] = tbl[(size_t)tok[bt]*D_ + c];
}

// conv tap: y[b,co,i] for kernel size K, given stride/pad (includes bias)
__device__ __forceinline__ float conv_val(const float* __restrict__ w,
                                          const float* __restrict__ bias,
                                          const float* __restrict__ emb,
                                          int b, int co, int i,
                                          int K, int stride, int pad){
  float acc = bias[co];
  int t0 = i*stride - pad;
  const float* wrow = w + (size_t)co * D_ * K;
  for(int kk=0; kk<K; ++kk){
    int t = t0 + kk;
    if(t < 0 || t >= L_) continue;
    const float* xr = emb + ((size_t)b*L_ + t)*D_;
    const float* wr = wrow + kk;
    float s = 0.f;
    for(int ci=0; ci<D_; ++ci) s += wr[(size_t)ci*K] * xr[ci];
    acc += s;
  }
  return acc;
}

// ---------------- kernel 2: conv streams -> A-fragment-packed bf16 --------
// layout: [sid][mtile(512)][kt(10)][lane(32)][e(16)]   (m = b*512+p)
__global__ void k_streams(const float* __restrict__ emb,
                          const float* w2,const float* c2,
                          const float* w4,const float* c4,
                          const float* w3,const float* c3,
                          const float* w6,const float* c6,
                          const float* w5,const float* c5,
                          __bf16* __restrict__ apack){
  int blk = blockIdx.x;             // 3*16*512
  int c   = threadIdx.x;            // 0..319  (= K index)
  int p   = blk & 511;
  int b   = (blk >> 9) & 15;
  int sid = blk >> 13;
  int m   = b*T_ + p;
  int mt = m >> 4, lo = m & 15;
  int kt = c >> 5, km = c & 31, hi, e;
  a_pat(km, hi, e);
  size_t dst = ((((size_t)sid*512 + mt)*KT_IH + kt)*32 + (hi*16 + lo))*16 + e;
  if(c >= D_){ apack[dst] = (__bf16)0.f; return; }
  float acc = 0.f;
  if(sid == 0){
    int i = inv_k2(p); if(i>=0) acc += conv_val(w2,c2,emb,b,c,i,2,1,0);
    i     = inv_k4(p); if(i>=0) acc += conv_val(w4,c4,emb,b,c,i,4,2,0);
  } else if(sid == 1){
    int i = inv_k3(p); if(i>=0) acc += conv_val(w3,c3,emb,b,c,i,3,3,1);
    i     = inv_k6(p); if(i>=0) acc += conv_val(w6,c6,emb,b,c,i,6,3,1);
  } else {
    int i = inv_k5(p); if(i>=0) acc += conv_val(w5,c5,emb,b,c,i,5,3,2);
    if(p & 1) acc += emb[((size_t)b*L_ + ((p-1)>>1))*D_ + c];
  }
  apack[dst] = (__bf16)acc;
}

// ---------------- weight packing (fp32 -> B-fragment bf16) ---------------
// B pattern within 32-chunk: same as A (k = 8*hi + e for e<8, 16 + 8*hi + (e-8) for e>=8)
// layout: [ntile(128)][kt][lane(32)][e(16)]
__global__ void k_pack_wih(const float* __restrict__ W, __bf16* __restrict__ out){
  int id = blockIdx.x*blockDim.x + threadIdx.x;   // 320*2048
  if(id >= KP_*G4_) return;
  int n = id & (G4_-1); int k = id >> 11;
  int nt = n >> 4, lo = n & 15;
  int kt = k >> 5, km = k & 31, hi = (km >> 3) & 1, e = (km & 7) + ((km >= 16) ? 8 : 0);
  size_t dst = (((size_t)nt*KT_IH + kt)*32 + (hi*16 + lo))*16 + e;
  out[dst] = (__bf16)((k < D_) ? W[(size_t)n*D_ + k] : 0.f);
}
__global__ void k_pack_whh(const float* __restrict__ W, __bf16* __restrict__ out){
  int id = blockIdx.x*blockDim.x + threadIdx.x;   // 512*2048
  if(id >= H_*G4_) return;
  int n = id & (G4_-1); int k = id >> 11;
  int nt = n >> 4, lo = n & 15;
  int kt = k >> 5, km = k & 31, hi = (km >> 3) & 1, e = (km & 7) + ((km >= 16) ? 8 : 0);
  size_t dst = (((size_t)nt*KT_HH + kt)*32 + (hi*16 + lo))*16 + e;
  out[dst] = (__bf16)W[(size_t)n*H_ + k];
}

// ---------------- kernel 3: input projection GEMM via WMMA ---------------
// wave computes a 16x64 strip (4 N tiles) sharing one A fragment per K step
__global__ void k_gemm_proj(const __bf16* __restrict__ Apack,
                            const __bf16* __restrict__ Bpack,
                            const float* __restrict__ bias,
                            float* __restrict__ C){
  const int wave = threadIdx.x >> 5;
  const int lane = threadIdx.x & 31;
  const int hi = lane >> 4, lo = lane & 15;
  int wg = blockIdx.x*8 + wave;       // 512 * 32 wave-groups
  int ng = wg & 31, tm = wg >> 5;
  v8f acc[4] = {{},{},{},{}};
  const __bf16* ap = Apack + (((size_t)tm*KT_IH)*32 + lane)*16;
  for(int kc=0; kc<KT_IH; ++kc){
    __builtin_prefetch(ap + 32*16, 0, 1);
    v16bf a = *(const v16bf*)ap;
    #pragma unroll
    for(int tt=0; tt<4; ++tt){
      const __bf16* bp = Bpack + ((((size_t)(ng*4+tt))*KT_IH + kc)*32 + lane)*16;
      v16bf bf = *(const v16bf*)bp;
      acc[tt] = __builtin_amdgcn_wmma_f32_16x16x32_bf16(false, a, false, bf,
                                                        (short)0, acc[tt], false, false);
    }
    ap += 32*16;
  }
  int m0 = tm*16;
  #pragma unroll
  for(int tt=0; tt<4; ++tt){
    int n = (ng*4+tt)*16 + lo;
    float bn = bias[n];
    #pragma unroll
    for(int r=0;r<8;++r)
      C[(size_t)(m0 + r + 8*hi)*G4_ + n] = acc[tt][r] + bn;
  }
}

// ---------------- kernel 4: LSTM recurrence (persistent, WMMA) -----------
// blockIdx.x: 0=fwd, 1=bwd.  Hs kept in packed A-fragment layout in LDS.
__global__ void __launch_bounds__(1024)
k_lstm(const float* __restrict__ proj,
       const __bf16* __restrict__ whhF, const __bf16* __restrict__ whhB,
       float* __restrict__ out){
  const int dir = blockIdx.x;
  const float* P = proj + (size_t)dir * (B_*T_) * G4_;
  const __bf16* Whh = dir ? whhB : whhF;
  extern __shared__ char smem[];
  float*  Gs = (float*)smem;                 // 16*2048 gate pre-activations
  float*  Cs = Gs + B_*G4_;                  // 16*512  cell state
  __bf16* Hs = (__bf16*)(Cs + B_*H_);        // 16*512  hidden, packed [kt][lane][16]
  for(int i=threadIdx.x; i<B_*H_; i+=blockDim.x){ Cs[i]=0.f; Hs[i]=(__bf16)0.f; }
  __syncthreads();
  const int wave = threadIdx.x >> 5;
  const int lane = threadIdx.x & 31;
  const int hi = lane >> 4, lo = lane & 15;
  for(int step=0; step<T_; ++step){
    int t = dir ? (T_-1-step) : step;
    // G = h @ Whh^T + proj[:, t, :]   (M=16 batch, N=2048, K=512)
    v8f acc[4] = {{},{},{},{}};
    for(int kc=0; kc<KT_HH; ++kc){
      v16bf a = *(const v16bf*)(Hs + ((size_t)kc*32 + lane)*16);
      #pragma unroll
      for(int tt=0; tt<4; ++tt){
        const __bf16* bp = Whh + ((((size_t)(wave*4+tt))*KT_HH + kc)*32 + lane)*16;
        __builtin_prefetch(bp + 32*16, 0, 1);
        v16bf bf = *(const v16bf*)bp;
        acc[tt] = __builtin_amdgcn_wmma_f32_16x16x32_bf16(false, a, false, bf,
                                                          (short)0, acc[tt], false, false);
      }
    }
    #pragma unroll
    for(int tt=0; tt<4; ++tt){
      int n = (wave*4+tt)*16 + lo;
      #pragma unroll
      for(int r=0;r<8;++r){
        int m = r + 8*hi;
        Gs[m*G4_ + n] = acc[tt][r] + P[((size_t)m*T_ + t)*G4_ + n];
      }
    }
    __syncthreads();
    // gates: i,f,g,o splits of 512; update c,h; publish packed h + output
    #pragma unroll
    for(int u=0; u<8; ++u){
      int idx = threadIdx.x + u*1024;      // 16*512 elements
      int b = idx >> 9, j = idx & 511;
      float iv = Gs[b*G4_ + j];
      float fv = Gs[b*G4_ + 512 + j];
      float gv = Gs[b*G4_ + 1024 + j];
      float ov = Gs[b*G4_ + 1536 + j];
      float cc = sigmoidf_(fv)*Cs[idx] + sigmoidf_(iv)*tanhf(gv);
      float hh = sigmoidf_(ov)*tanhf(cc);
      Cs[idx] = cc;
      int kt = j >> 5, km = j & 31, phi, pe;
      a_pat(km, phi, pe);
      Hs[((size_t)kt*32 + (phi*16 + b))*16 + pe] = (__bf16)hh;
      out[((size_t)b*T_ + t)*1024 + (size_t)dir*H_ + j] = hh;
    }
    __syncthreads();
  }
}

// ---------------- kernel 5: mix + crunch + MLP layer 1 (silu) ------------
__global__ void k_mlp1(const float* __restrict__ u, const float* __restrict__ m,
                       const float* __restrict__ l, const float* __restrict__ mw,
                       const float* __restrict__ W1, const float* __restrict__ b1,
                       float* __restrict__ h16){
  int id = blockIdx.x*blockDim.x + threadIdx.x;   // 16*256*16
  if(id >= B_*256*16) return;
  int j = id & 15, s = (id>>4) & 255, b = id >> 12;
  float e0=expf(mw[0]), e1=expf(mw[1]), e2=expf(mw[2]), e3=expf(mw[3]);
  float inv = 1.f/(e0+e1+e2+e3);
  float w0=e0*inv, w1=e1*inv, w2=e2*inv, w3=e3*inv;
  float au = 0.25f*(w0+w1+w2+w3);
  float am = 0.25f*(w0+w1+w3);
  float al = 0.25f*(w2+w3);
  float acc = b1[j];
  const float* Wr = W1 + (size_t)j*G4_;
  for(int q=0; q<G4_; ++q){
    int tt = 2*s + (q>>10);            // even rows then odd rows
    int qp = q & 1023;
    size_t fi = ((size_t)b*T_ + tt)*1024 + qp;
    float f = au*u[fi] + am*m[fi] + al*l[fi];
    acc += Wr[q]*f;
  }
  h16[id] = acc / (1.f + expf(-acc));  // silu
}

// ---------------- kernel 6: logits + softmax over time axis --------------
__global__ void k_head2(const float* __restrict__ h16, const float* __restrict__ W2,
                        const float* __restrict__ b2, float* __restrict__ outp){
  int b = blockIdx.x >> 1, cls = blockIdx.x & 1;
  int s = threadIdx.x;                 // 0..255
  float acc = b2[cls];
  const float* hr = h16 + ((size_t)b*256 + s)*16;
  #pragma unroll
  for(int j=0;j<16;++j) acc += hr[j]*W2[cls*16 + j];
  __shared__ float red[256];
  red[s] = acc; __syncthreads();
  for(int off=128; off>0; off>>=1){ if(s<off) red[s]=fmaxf(red[s],red[s+off]); __syncthreads(); }
  float mx = red[0]; __syncthreads();
  float ex = expf(acc - mx);
  red[s] = ex; __syncthreads();
  for(int off=128; off>0; off>>=1){ if(s<off) red[s]+=red[s+off]; __syncthreads(); }
  float denom = red[0];
  outp[((size_t)b*256 + s)*2 + cls] = ex/denom;
}

// ---------------- launcher ----------------
extern "C" void kernel_launch(void* const* d_in, const int* in_sizes, int n_in,
                              void* d_out, int out_size, void* d_ws, size_t ws_size,
                              hipStream_t stream){
  (void)in_sizes; (void)n_in; (void)out_size; (void)ws_size;
  const int*   tokens = (const int*)  d_in[0];
  const float* table  = (const float*)d_in[1];
  const float* mixw = (const float*)d_in[30];
  const float* W1   = (const float*)d_in[31];
  const float* b1   = (const float*)d_in[32];
  const float* W2   = (const float*)d_in[33];
  const float* b2   = (const float*)d_in[34];

  // ---- workspace carve-up ----
  char* w = (char*)d_ws;
  float*  emb     = (float*)w;                      w += (size_t)B_*L_*D_*4;          // 4.9 MB
  __bf16* streams = (__bf16*)w;                     w += (size_t)3*B_*T_*KP_*2;       // 15.7 MB
  __bf16* wihT    = (__bf16*)w;                     w += (size_t)6*KP_*G4_*2;         // 7.9 MB
  __bf16* whhT    = (__bf16*)w;                     w += (size_t)6*H_*G4_*2;          // 12.6 MB
  float*  proj    = (float*)w;                      w += (size_t)2*B_*T_*G4_*4;       // 134 MB
  float*  hs      = (float*)w;                      w += (size_t)3*B_*T_*1024*4;      // 100.7 MB
  float*  h16     = (float*)w;                      w += (size_t)B_*256*16*4;

  // 1) embedding gather
  k_embed<<<(B_*L_*D_ + 255)/256, 256, 0, stream>>>(tokens, table, emb);

  // 2) conv streams -> packed A fragments
  k_streams<<<3*B_*T_, KP_, 0, stream>>>(emb,
      (const float*)d_in[2], (const float*)d_in[3],
      (const float*)d_in[4], (const float*)d_in[5],
      (const float*)d_in[6], (const float*)d_in[7],
      (const float*)d_in[8], (const float*)d_in[9],
      (const float*)d_in[10],(const float*)d_in[11],
      streams);

  // 3) pack all LSTM weights into B-fragment layouts
  for(int lv=0; lv<3; ++lv){
    for(int d=0; d<2; ++d){
      int base = 12 + lv*6 + d*3;
      int slot = lv*2 + d;
      k_pack_wih<<<(KP_*G4_ + 255)/256, 256, 0, stream>>>(
          (const float*)d_in[base+0], wihT + (size_t)slot*KP_*G4_);
      k_pack_whh<<<(H_*G4_ + 255)/256, 256, 0, stream>>>(
          (const float*)d_in[base+1], whhT + (size_t)slot*H_*G4_);
    }
  }

  // 4) per level: input-projection GEMMs (WMMA) then persistent BiLSTM
  size_t smem = (size_t)B_*G4_*4 + (size_t)B_*H_*4 + (size_t)B_*H_*2; // 176 KB
  for(int lv=0; lv<3; ++lv){
    const __bf16* A = streams + (size_t)lv*B_*T_*KP_;
    for(int d=0; d<2; ++d){
      int base = 12 + lv*6 + d*3;
      int slot = lv*2 + d;
      k_gemm_proj<<<(512*32)/8, 256, 0, stream>>>(
          A, wihT + (size_t)slot*KP_*G4_,
          (const float*)d_in[base+2],
          proj + (size_t)d*B_*T_*G4_);
    }
    k_lstm<<<2, 1024, smem, stream>>>(
        proj,
        whhT + (size_t)(lv*2+0)*H_*G4_,
        whhT + (size_t)(lv*2+1)*H_*G4_,
        hs + (size_t)lv*B_*T_*1024);
  }

  // 5) head
  k_mlp1<<<(B_*256*16 + 255)/256, 256, 0, stream>>>(
      hs, hs + (size_t)1*B_*T_*1024, hs + (size_t)2*B_*T_*1024,
      mixw, W1, b1, h16);
  k_head2<<<B_*2, 256, 0, stream>>>(h16, W2, b2, (float*)d_out);
}
